// PatchAttentionNet_64536178590152
// MI455X (gfx1250) — matrix operand order repaired and checked by reading, not compiled, not run on hardware
//
#include <hip/hip_runtime.h>
#include <hip/hip_bf16.h>

// Problem constants (from reference): B=16, C=64, G=256, P=4096, W=1024, H=256
#define BB 16
#define CC 64
#define GG 256
#define PP 4096
#define WW 1024
#define HH 256
#define WQ 256   // W/4
// D = C+G+WQ = 576 ; W1 rows: [0,64)=W1c, [64,320)=W1g, [320,576)=W1w

typedef __attribute__((ext_vector_type(16))) __bf16    v16bf;
typedef __attribute__((ext_vector_type(8)))  float     v8f;
typedef __attribute__((ext_vector_type(4)))  float     v4f;
typedef __attribute__((ext_vector_type(8)))  unsigned  v8u;

__device__ __forceinline__ unsigned short f2bf(float f) {
    unsigned u = __builtin_bit_cast(unsigned, f);
    unsigned r = u + 0x7FFFu + ((u >> 16) & 1u);   // round-to-nearest-even
    return (unsigned short)(r >> 16);
}

// ---------------------------------------------------------------------------
// Prep: convert Ww (WxWQ, row-major) -> bf16 column-major [q][k], and
//       W1w (= W1[320:576,:], q x h row-major) -> bf16 column-major [h][q].
// Column-major bf16 lets each WMMA lane fetch its 16 contiguous K values
// (32 bytes) with a single wide global load.
// ---------------------------------------------------------------------------
__global__ void prep_bf16(const float* __restrict__ Ww, const float* __restrict__ W1,
                          unsigned short* __restrict__ WwB, unsigned short* __restrict__ W1wB) {
    int i = blockIdx.x * 256 + threadIdx.x;
    if (i < WW * WQ) {                       // WwB[q*1024 + k] = Ww[k*256 + q]
        int q = i >> 10, k = i & (WW - 1);
        WwB[i] = f2bf(Ww[k * WQ + q]);
    } else {
        int j = i - WW * WQ;
        if (j < WQ * HH) {                   // W1wB[h*256 + q] = W1[(320+q)*256 + h]
            int h = j >> 8, q = j & 255;
            W1wB[j] = f2bf(W1[(CC + GG + q) * HH + h]);
        }
    }
}

// ---------------------------------------------------------------------------
// Tiny dense layers: c = cx@Wc+bc ; g = gx@Wg+bg   (one block per batch row)
// ---------------------------------------------------------------------------
__global__ void small_cg(const float* __restrict__ cx, const float* __restrict__ gx,
                         const float* __restrict__ Wc, const float* __restrict__ bc,
                         const float* __restrict__ Wg, const float* __restrict__ bg,
                         float* __restrict__ cbuf, float* __restrict__ gbuf) {
    int b = blockIdx.x, t = threadIdx.x;
    if (t < CC) {
        float a = bc[t];
        for (int j = 0; j < CC; ++j) a += cx[b * CC + j] * Wc[j * CC + t];
        cbuf[b * CC + t] = a;
    }
    float a = bg[t];
    for (int j = 0; j < GG; ++j) a += gx[b * GG + j] * Wg[j * GG + t];
    gbuf[b * GG + t] = a;
}

// z[b][h] = b1[h] + c[b]@W1c[:,h] + g[b]@W1g[:,h]  (broadcast term of h)
__global__ void small_z(const float* __restrict__ cbuf, const float* __restrict__ gbuf,
                        const float* __restrict__ W1, const float* __restrict__ b1,
                        float* __restrict__ zbuf) {
    int b = blockIdx.x, t = threadIdx.x;
    float a = b1[t];
    for (int k = 0; k < CC; ++k) a += cbuf[b * CC + k] * W1[k * HH + t];
    for (int k = 0; k < GG; ++k) a += gbuf[b * GG + k] * W1[(CC + k) * HH + t];
    zbuf[b * HH + t] = a;
}

__global__ void zero_out(float* __restrict__ out) {
    out[blockIdx.x * 256 + threadIdx.x] = 0.0f;
}

// ---------------------------------------------------------------------------
// Main fused kernel: one workgroup (8 waves) per 16-row P tile.
//   GEMM1: w16x256 = bf16(wx tile 16x1024) @ WwB   -> +bw -> wbuf (fp32) + LDS bf16
//   GEMM2: h16x256 = relu(z + wtile @ W1wB)        -> scores via h@W2+b2
// A fragment order used here (contiguous k, 16x32): lanes 0-15 row M=lane, halfword pairs
//   elems 0..15 = K 0..15; lanes 16-31 same rows, K 16..31.
// B fragment order used here (matches A, 32x16): lane<16: col N=lane, K = K0..K0+15;
//   lane>=16: col N=lane-16, K = K0+16..K0+31 (contiguous in column-major).
// ---------------------------------------------------------------------------
__global__ __launch_bounds__(256) void patch_main(
        const float* __restrict__ wx, const float* __restrict__ bw,
        const float* __restrict__ W2, const float* __restrict__ b2,
        const float* __restrict__ zbuf,
        const unsigned short* __restrict__ WwB, const unsigned short* __restrict__ W1wB,
        float* __restrict__ wbuf, float* __restrict__ sbuf) {
    __shared__ unsigned short Alds[16 * WW];   // 32 KB: bf16 wx tile, K-contig rows
    __shared__ unsigned short Wlds[16 * WQ];   // 8 KB : bf16 w tile for GEMM2
    __shared__ float scoreLds[16];

    const int tid = threadIdx.x;
    const int b  = blockIdx.x >> 8;            // 256 tiles per batch
    const int p0 = (blockIdx.x & 255) * 16;

    // ---- stage A: 16x1024 fp32 -> bf16 LDS. wx is a 256 MB single-use
    // stream (> 192 MB L2): use non-temporal loads so weights + wbuf stay
    // L2-resident for the downstream passes. ----
    const float* wrow = wx + ((size_t)(b * PP + p0)) * WW;
    #pragma unroll 4
    for (int i = 0; i < 16; ++i) {
        v4f f = __builtin_nontemporal_load((const v4f*)(wrow + (size_t)i * WW + tid * 4));
        unsigned lo  = (unsigned)f2bf(f[0]) | ((unsigned)f2bf(f[1]) << 16);
        unsigned hi2 = (unsigned)f2bf(f[2]) | ((unsigned)f2bf(f[3]) << 16);
        *(uint2*)(Alds + i * WW + tid * 4) = make_uint2(lo, hi2);
    }
    if (tid < 16) scoreLds[tid] = b2[0];
    __syncthreads();

    const int lane = tid & 31, wv = tid >> 5;
    const int lm = lane & 15, hi = lane >> 4;
    const int n0 = (wv * 2) * 16 + lm, n1 = n0 + 16;

    // ---- GEMM1: acc = A(16x1024) @ Ww(1024x256), two N tiles per wave ----
    v8f acc0 = {}; v8f acc1 = {};
    {
        const unsigned* arow = (const unsigned*)Alds + lm * (WW / 2);
        const unsigned short* bc0 = WwB + (size_t)n0 * WW + hi * 16;
        const unsigned short* bc1 = WwB + (size_t)n1 * WW + hi * 16;
        #pragma unroll 2
        for (int K0 = 0; K0 < WW; K0 += 32) {
            v8u au;
            int base = (K0 >> 1) + hi * 8;
            au[0] = arow[base + 0]; au[1] = arow[base + 1];
            au[2] = arow[base + 2]; au[3] = arow[base + 3];
            au[4] = arow[base + 4]; au[5] = arow[base + 5];
            au[6] = arow[base + 6]; au[7] = arow[base + 7];
            v16bf a  = __builtin_bit_cast(v16bf, au);
            v16bf b0 = __builtin_bit_cast(v16bf, *(const v8u*)(bc0 + K0));
            v16bf b1v = __builtin_bit_cast(v16bf, *(const v8u*)(bc1 + K0));
            acc0 = __builtin_amdgcn_wmma_f32_16x16x32_bf16(false, a, false, b0,  (short)0, acc0, false, false);
            acc1 = __builtin_amdgcn_wmma_f32_16x16x32_bf16(false, a, false, b1v, (short)0, acc1, false, false);
        }
    }

    // ---- +bw ; spill fp32 w tile; stage bf16 copy for GEMM2 ----
    {
        float bw0 = bw[n0], bw1 = bw[n1];
        #pragma unroll
        for (int r = 0; r < 8; ++r) {
            int m = r + hi * 8;                         // C/D layout: VGPR r -> M
            float w0 = acc0[r] + bw0, w1 = acc1[r] + bw1;
            size_t ro = ((size_t)(b * PP + p0 + m)) * WQ;
            wbuf[ro + n0] = w0; wbuf[ro + n1] = w1;
            Wlds[m * WQ + n0] = f2bf(w0);
            Wlds[m * WQ + n1] = f2bf(w1);
        }
    }
    __syncthreads();

    // ---- GEMM2: h = relu(z + w @ W1w); scores += h*W2 ----
    v8f h0 = {}; v8f h1 = {};
    {
        const unsigned* arow = (const unsigned*)Wlds + lm * (WQ / 2);
        const unsigned short* bc0 = W1wB + (size_t)n0 * WQ + hi * 16;
        const unsigned short* bc1 = W1wB + (size_t)n1 * WQ + hi * 16;
        #pragma unroll
        for (int K0 = 0; K0 < WQ; K0 += 32) {
            v8u au;
            int base = (K0 >> 1) + hi * 8;
            au[0] = arow[base + 0]; au[1] = arow[base + 1];
            au[2] = arow[base + 2]; au[3] = arow[base + 3];
            au[4] = arow[base + 4]; au[5] = arow[base + 5];
            au[6] = arow[base + 6]; au[7] = arow[base + 7];
            v16bf a  = __builtin_bit_cast(v16bf, au);
            v16bf b0 = __builtin_bit_cast(v16bf, *(const v8u*)(bc0 + K0));
            v16bf b1v = __builtin_bit_cast(v16bf, *(const v8u*)(bc1 + K0));
            h0 = __builtin_amdgcn_wmma_f32_16x16x32_bf16(false, a, false, b0,  (short)0, h0, false, false);
            h1 = __builtin_amdgcn_wmma_f32_16x16x32_bf16(false, a, false, b1v, (short)0, h1, false, false);
        }
    }
    {
        float z0 = zbuf[b * HH + n0], z1 = zbuf[b * HH + n1];
        float w20 = W2[n0], w21 = W2[n1];
        #pragma unroll
        for (int r = 0; r < 8; ++r) {
            // butterfly-reduce across the 16 lanes of each half-wave: lanes
            // 0-15 hold row m=r, lanes 16-31 hold m=r+8 (xor masks stay in-half)
            float hv = fmaxf(h0[r] + z0, 0.0f) * w20 + fmaxf(h1[r] + z1, 0.0f) * w21;
            hv += __shfl_xor(hv, 1, 32);
            hv += __shfl_xor(hv, 2, 32);
            hv += __shfl_xor(hv, 4, 32);
            hv += __shfl_xor(hv, 8, 32);
            if (lm == 0) atomicAdd(&scoreLds[r + hi * 8], hv);
        }
    }
    __syncthreads();
    if (tid < 16) sbuf[b * PP + p0 + tid] = scoreLds[tid];
}

// ---------------------------------------------------------------------------
// Softmax over P + attention-weighted sum. Grid (B, 8): each block redundantly
// computes the max / sum-exp over all 4096 scores (16 KB, L2-hot), then
// accumulates its 512-row slice of sum_p attn*w into d_out via fp32 atomics.
// ---------------------------------------------------------------------------
__global__ __launch_bounds__(256) void softmax_reduce(
        const float* __restrict__ sbuf, const float* __restrict__ wbuf,
        float* __restrict__ out) {
    __shared__ float red[256];
    __shared__ float pr[512];
    const int b = blockIdx.x, t = threadIdx.x;
    const float* s = sbuf + b * PP;

    float m = -3.402823466e38f;
    for (int j = t; j < PP; j += 256) m = fmaxf(m, s[j]);
    red[t] = m; __syncthreads();
    for (int off = 128; off > 0; off >>= 1) {
        if (t < off) red[t] = fmaxf(red[t], red[t + off]);
        __syncthreads();
    }
    const float gmax = red[0]; __syncthreads();

    float sum = 0.0f;
    for (int j = t; j < PP; j += 256) sum += __expf(s[j] - gmax);
    red[t] = sum; __syncthreads();
    for (int off = 128; off > 0; off >>= 1) {
        if (t < off) red[t] += red[t + off];
        __syncthreads();
    }
    const float invZ = 1.0f / red[0]; __syncthreads();

    const int pz = blockIdx.y * 512;
    pr[t]       = __expf(s[pz + t] - gmax);
    pr[t + 256] = __expf(s[pz + 256 + t] - gmax);
    __syncthreads();

    float acc = 0.0f;
    const float* wb = wbuf + ((size_t)(b * PP + pz)) * WQ + t;
    #pragma unroll 8
    for (int i = 0; i < 512; ++i) acc += pr[i] * wb[(size_t)i * WQ];
    atomicAdd(&out[b * WQ + t], acc * invZ);
}

// ---------------------------------------------------------------------------
extern "C" void kernel_launch(void* const* d_in, const int* in_sizes, int n_in,
                              void* d_out, int out_size, void* d_ws, size_t ws_size,
                              hipStream_t stream) {
    (void)in_sizes; (void)n_in; (void)out_size; (void)ws_size;
    const float* cx = (const float*)d_in[0];
    const float* gx = (const float*)d_in[1];
    const float* wx = (const float*)d_in[2];
    const float* Wc = (const float*)d_in[3];
    const float* bc = (const float*)d_in[4];
    const float* Wg = (const float*)d_in[5];
    const float* bg = (const float*)d_in[6];
    const float* Ww = (const float*)d_in[7];
    const float* bw = (const float*)d_in[8];
    const float* W1 = (const float*)d_in[9];
    const float* b1 = (const float*)d_in[10];
    const float* W2 = (const float*)d_in[11];
    const float* b2 = (const float*)d_in[12];
    float* out = (float*)d_out;

    char* ws = (char*)d_ws;
    size_t off = 0;
    auto carve = [&](size_t bytes) { size_t o = off; off = (off + bytes + 255) & ~(size_t)255; return o; };
    unsigned short* WwB  = (unsigned short*)(ws + carve((size_t)WW * WQ * 2)); // 512 KB
    unsigned short* W1wB = (unsigned short*)(ws + carve((size_t)WQ * HH * 2)); // 128 KB
    float* cbuf = (float*)(ws + carve((size_t)BB * CC * 4));
    float* gbuf = (float*)(ws + carve((size_t)BB * GG * 4));
    float* zbuf = (float*)(ws + carve((size_t)BB * HH * 4));
    float* sbuf = (float*)(ws + carve((size_t)BB * PP * 4));                   // 256 KB
    float* wbuf = (float*)(ws + carve((size_t)BB * PP * WQ * 4));              // 64 MB

    prep_bf16<<<(WW * WQ + WQ * HH + 255) / 256, 256, 0, stream>>>(Ww, W1, WwB, W1wB);
    small_cg<<<BB, 256, 0, stream>>>(cx, gx, Wc, bc, Wg, bg, cbuf, gbuf);
    small_z<<<BB, 256, 0, stream>>>(cbuf, gbuf, W1, b1, zbuf);
    zero_out<<<BB, 256, 0, stream>>>(out);   // BB*256 == BB*WQ outputs
    patch_main<<<BB * (PP / 16), 256, 0, stream>>>(wx, bw, W2, b2, zbuf, WwB, W1wB, wbuf, sbuf);
    softmax_reduce<<<dim3(BB, 8), 256, 0, stream>>>(sbuf, wbuf, out);
}
